// Ortho_GCNII_37984690765993
// MI455X (gfx1250) — matrix operand order repaired and checked by reading, not compiled, not run on hardware
//
#include <hip/hip_runtime.h>
#include <hip/hip_bf16.h>
#include <math.h>

// ---------------------------------------------------------------------------
// Ortho-GCNII forward for MI455X (gfx1250, wave32, WMMA).
//   out = theta * (support @ t) + (1-theta) * support
//   support = 0.9 * (A_sparse @ input) + 0.1 * h0
//   t = group Newton-Schulz orthogonalization of 0.5*W + 0.5*I  (f32 WMMA)
// Big GEMM: V_WMMA_F32_16X16X32_F16 + async global->LDS double-buffered A.
// Ortho path: V_WMMA_F32_16X16X4_F32, branch-free templated inner loop.
// ---------------------------------------------------------------------------

typedef __attribute__((ext_vector_type(16))) _Float16 v16h;
typedef __attribute__((ext_vector_type(8)))  float    v8f;
typedef __attribute__((ext_vector_type(2)))  float    v2f;
typedef __attribute__((ext_vector_type(4)))  int      v4i;

typedef __attribute__((address_space(1))) v4i as1_v4i;
typedef __attribute__((address_space(3))) v4i as3_v4i;

#define DDIM 512
#define NGROUP 2
#define GD 256           // DDIM / NGROUP
#define EPSV 1e-5f

#if defined(__has_builtin)
#if __has_builtin(__builtin_amdgcn_global_load_async_to_lds_b128) && \
    __has_builtin(__builtin_amdgcn_s_wait_asynccnt)
#define HAVE_ASYNC_COPY 1
#endif
#endif
#ifndef HAVE_ASYNC_COPY
#define HAVE_ASYNC_COPY 0
#endif

union U16 { uint4 u[2]; v16h h; };

// ---------------------------------------------------------------- utilities
__global__ void zero_k(float4* __restrict__ p, long n4) {
  long i = (long)blockIdx.x * blockDim.x + threadIdx.x;
  long st = (long)gridDim.x * blockDim.x;
  float4 z = {0.f, 0.f, 0.f, 0.f};
  for (; i < n4; i += st) p[i] = z;
}

// support = 0.9*hi + 0.1*h0 (in place over hi), plus f16 copy for the GEMM A.
__global__ void support_k(float4* __restrict__ hi, const float4* __restrict__ h0,
                          uint2* __restrict__ s16, long n4) {
  long i = (long)blockIdx.x * blockDim.x + threadIdx.x;
  long st = (long)gridDim.x * blockDim.x;
  for (; i < n4; i += st) {
    float4 a = hi[i], b = h0[i];
    float4 r;
    r.x = 0.9f*a.x + 0.1f*b.x; r.y = 0.9f*a.y + 0.1f*b.y;
    r.z = 0.9f*a.z + 0.1f*b.z; r.w = 0.9f*a.w + 0.1f*b.w;
    hi[i] = r;
    union { _Float16 h[4]; uint2 u; } pk;
    pk.h[0] = (_Float16)r.x; pk.h[1] = (_Float16)r.y;
    pk.h[2] = (_Float16)r.z; pk.h[3] = (_Float16)r.w;
    s16[i] = pk.u;
  }
}

// ------------------------------------------------------- SpMM via f32 atomics
// one block per edge; 256 threads cover D=512 coalesced; HW f32 atomic adds
// land in L2 (whole 100 MB hi array is L2-resident on the 192 MB MI455X L2).
__global__ void spmm_k(const float* __restrict__ x, const int* __restrict__ row,
                       const int* __restrict__ col, const float* __restrict__ val,
                       float* __restrict__ hi) {
  int e = blockIdx.x;
  int r = row[e], c = col[e];
  float v = val[e];
  const float* src = x + (long)c * DDIM;
  float* dst = hi + (long)r * DDIM;
  int t = threadIdx.x;
  unsafeAtomicAdd(&dst[t],       v * src[t]);
  unsafeAtomicAdd(&dst[t + 256], v * src[t + 256]);
}

// ------------------------------------------- Zc = (0.5*W + 0.5*I) - row mean
__global__ void zc_k(const float* __restrict__ w, float* __restrict__ Zc) {
  int r = blockIdx.x;            // 0..511
  int t = threadIdx.x;           // 256
  __shared__ float red[256];
  float s = 0.f;
  for (int c = t; c < DDIM; c += 256) {
    float we = 0.5f * w[(long)r*DDIM + c] + ((c == r) ? 0.5f : 0.f);
    s += we;
  }
  red[t] = s; __syncthreads();
  for (int o = 128; o > 0; o >>= 1) { if (t < o) red[t] += red[t+o]; __syncthreads(); }
  float mean = red[0] * (1.0f / DDIM);
  for (int c = t; c < DDIM; c += 256) {
    float we = 0.5f * w[(long)r*DDIM + c] + ((c == r) ? 0.5f : 0.f);
    Zc[(long)r*DDIM + c] = we - mean;
  }
}

// ------------------------------- generic small batched f32 WMMA GEMM (wave32)
// Each wave computes a 16x64 C strip (4 accs) with V_WMMA_F32_16X16X4_F32.
// A frag (16x4): lane<16 -> row=lane, K{0,1}; lane>=16 -> K{2,3}. (ISA 7.12.2)
// B frag (4x16): lane<16 -> col=lane, K{0,1}; lane>=16 -> K{2,3}.
// TRANSB=1: Bop stored [n][k] (contiguous v2f loads) -- all Newton-Schulz
// factors (B1/P/Q/Sn) are symmetric so NT is exact.
// C = (alpha*acc + beta*aux + eps*diag) * scale[g].
template <int TRANSB>
__global__ void gemm32_k(const float* __restrict__ A, int lda, long sA,
                         const float* __restrict__ B, int ldb, long sB,
                         float* __restrict__ C, int ldc, long sC,
                         const float* __restrict__ aux, int ldaux, long sAux,
                         const float* __restrict__ scalep, int scaleStride,
                         int M, int Ncols, int K,
                         float alpha, float beta, int diagEps) {
  int wave = threadIdx.x >> 5;
  int lane = threadIdx.x & 31;
  int lm = lane & 15, lh = lane >> 4;
  int tilesN = Ncols >> 6;                     // 64-wide strips
  int tileId = blockIdx.x * 8 + wave;
  if (tileId >= (M >> 4) * tilesN) return;     // uniform per wave: EXEC stays full
  int tm = tileId / tilesN, tn = tileId - tm * tilesN;
  int g = blockIdx.z;
  const float* Ag = A + (long)g * sA;
  const float* Bg = B + (long)g * sB;
  int row = tm * 16 + lm;
  int col0 = tn * 64 + lm;
  v8f acc[4];
#pragma unroll
  for (int j = 0; j < 4; j++) acc[j] = (v8f){};

  for (int k = 0; k < K; k += 8) {
#pragma unroll
    for (int kk = 0; kk < 8; kk += 4) {
      int ka = k + kk + 2 * lh;
      v2f a = *(const v2f*)(Ag + (long)row * lda + ka);
#pragma unroll
      for (int j = 0; j < 4; j++) {
        int col = col0 + j * 16;
        v2f b;
        if (TRANSB) {
          b = *(const v2f*)(Bg + (long)col * ldb + ka);
        } else {
          b.x = Bg[(long)ka * ldb + col];
          b.y = Bg[(long)(ka + 1) * ldb + col];
        }
        acc[j] = __builtin_amdgcn_wmma_f32_16x16x4_f32(false, a, false, b,
                                                       (short)0, acc[j], false, false);
      }
    }
  }
  float scale = scalep ? scalep[g * scaleStride] : 1.0f;
  float* Cg = C + (long)g * sC;
  const float* Xg = aux ? (aux + (long)g * sAux) : nullptr;
#pragma unroll
  for (int j = 0; j < 4; j++) {
    int col = col0 + j * 16;
#pragma unroll
    for (int i = 0; i < 8; i++) {
      int r = tm * 16 + (lh ? (i + 8) : i);
      float v = alpha * acc[j][i];
      if (Xg) v += beta * Xg[(long)r * ldaux + col];
      if (diagEps && r == col) v += EPSV;
      Cg[(long)r * ldc + col] = v * scale;
    }
  }
}

// --------------------------------------- per-group Frobenius norm of S (+eps)
__global__ void norm_k(const float* __restrict__ S, float* __restrict__ gnorm) {
  int g = blockIdx.x;
  const float* Sg = S + (long)g * GD * GD;
  __shared__ float red[256];
  float s = 0.f;
  for (int i = threadIdx.x; i < GD * GD; i += 256) { float v = Sg[i]; s += v * v; }
  red[threadIdx.x] = s; __syncthreads();
  for (int o = 128; o > 0; o >>= 1) {
    if (threadIdx.x < o) red[threadIdx.x] += red[threadIdx.x + o];
    __syncthreads();
  }
  if (threadIdx.x == 0) {
    float nrm = sqrtf(red[0]);
    gnorm[2 * g]     = 1.0f / nrm;   // S normalization
    gnorm[2 * g + 1] = rsqrtf(nrm);  // final W scale
  }
}

// Sn = S/|S| (in place); B1 = 1.5I - 0.5*Sn  (first NS iteration with B=I)
__global__ void snb1_k(float* __restrict__ S, float* __restrict__ B1,
                       const float* __restrict__ gnorm) {
  int idx = blockIdx.x * blockDim.x + threadIdx.x;
  if (idx >= NGROUP * GD * GD) return;
  int g = idx / (GD * GD);
  int rc = idx - g * GD * GD;
  int r = rc >> 8, c = rc & 255;
  float sn = S[idx] * gnorm[2 * g];
  S[idx] = sn;
  B1[idx] = ((r == c) ? 1.5f : 0.f) - 0.5f * sn;
}

// ------------------------- pack t (f32 row-major) into f16 WMMA B fragments
// B frag k order used here (16-bit 32x16, contiguous, same as A reads): lanes 0-15 hold K=0..15
// (dword j = K{2j,2j+1}), lanes 16-31 hold K=16..31; col = lane%16.
// Packed stream: [kt][ntile][lane][8 dwords] -> each lane reads 32 contig B.
__global__ void pack_k(const float* __restrict__ T, unsigned* __restrict__ P) {
  int tid = blockIdx.x * blockDim.x + threadIdx.x;     // DDIM*DDIM/2 dwords
  if (tid >= DDIM * DDIM / 2) return;
  int j    = tid & 7;
  int lane = (tid >> 3) & 31;
  int ntg  = (tid >> 8) & 31;
  int kt   = tid >> 13;
  int colc = ntg * 16 + (lane & 15);
  int k0   = kt * 32 + (lane >> 4) * 16 + 2 * j;
  union { _Float16 h[2]; unsigned u; } pk;
  pk.h[0] = (_Float16)T[(long)k0 * DDIM + colc];
  pk.h[1] = (_Float16)T[(long)(k0 + 1) * DDIM + colc];
  P[tid] = pk.u;
}

// --------------------------- big GEMM: out = theta*(support@t) + (1-theta)*s
// block: 256 thr = 8 waves (2M x 4N), block tile 64 x 256, wave tile 32 x 64,
// K stepped by 32 with V_WMMA_F32_16X16X32_F16. A (pre-converted f16) is
// staged into LDS with double-buffered GLOBAL_LOAD_ASYNC_TO_LDS_B128
// (ASYNCcnt + barrier), row stride padded to 40 halves for conflict-free
// ds_load_b128 fragment reads.
__global__ __launch_bounds__(256) void biggemm_k(const float* __restrict__ Sup,
                                                 const _Float16* __restrict__ Sup16,
                                                 const unsigned* __restrict__ Bp,
                                                 const int* __restrict__ lptr,
                                                 float* __restrict__ Out, int Nrows) {
  __shared__ __align__(16) _Float16 Atile[2][64 * 40];
  int t = threadIdx.x;
  int wave = t >> 5, lane = t & 31;
  int lm = lane & 15, lh = lane >> 4;
  int wm = wave >> 2, wn = wave & 3;
  long rowBase = (long)blockIdx.x * 64;
  int colBase = blockIdx.y * 256;

  v8f acc[2][4];
#pragma unroll
  for (int mi = 0; mi < 2; mi++)
#pragma unroll
    for (int ni = 0; ni < 4; ni++) acc[mi][ni] = (v8f){};

  // staging: each thread moves one 16 B chunk (8 halves) of the 64x32 A tile
  int srow = t >> 2;
  int schunk = (t & 3) * 8;
  long grow = rowBase + srow;
  if (grow >= Nrows) grow = 0;                 // clamp, stores are guarded
  const _Float16* asrc = Sup16 + grow * DDIM + schunk;
  _Float16* adst[2] = { &Atile[0][srow * 40 + schunk], &Atile[1][srow * 40 + schunk] };
  const uint4* bp4 = (const uint4*)Bp;

  auto stage = [&](int kt, int buf) {
#if HAVE_ASYNC_COPY
    __builtin_amdgcn_global_load_async_to_lds_b128(
        (as1_v4i*)(asrc + kt * 32), (as3_v4i*)adst[buf], 0, 0);
#else
    *(uint4*)adst[buf] = *(const uint4*)(asrc + kt * 32);
#endif
  };
  auto wait_stage = [&]() {
#if HAVE_ASYNC_COPY
    __builtin_amdgcn_s_wait_asynccnt(0);
#endif
    __syncthreads();
  };

  stage(0, 0);
  wait_stage();

  for (int kt = 0; kt < 16; kt++) {
    int cur = kt & 1;
    if (kt < 15) stage(kt + 1, cur ^ 1);       // prefetch next tile (async)

    const _Float16* At = Atile[cur];
    U16 afr[2];
#pragma unroll
    for (int mi = 0; mi < 2; mi++) {
      int rl = wm * 32 + mi * 16 + lm;
      const _Float16* ap = &At[rl * 40 + lh * 16];
      afr[mi].u[0] = *(const uint4*)(ap);
      afr[mi].u[1] = *(const uint4*)(ap + 8);
    }
#pragma unroll
    for (int ni = 0; ni < 4; ni++) {
      int ntg = (colBase >> 4) + wn * 4 + ni;
      long b2 = (((long)kt * 32 + ntg) * 32 + lane) * 2;
      U16 bfr;
      bfr.u[0] = bp4[b2];
      bfr.u[1] = bp4[b2 + 1];
#pragma unroll
      for (int mi = 0; mi < 2; mi++)
        acc[mi][ni] = __builtin_amdgcn_wmma_f32_16x16x32_f16(
            false, afr[mi].h, false, bfr.h, (short)0, acc[mi][ni], false, false);
    }
    if (kt < 15) wait_stage();                 // next tile landed + all reads done
  }

  float lv = (float)lptr[0];
  float theta = logf(0.5f / lv + 1.0f);
  float omt = 1.0f - theta;
#pragma unroll
  for (int mi = 0; mi < 2; mi++)
#pragma unroll
    for (int ni = 0; ni < 4; ni++)
#pragma unroll
      for (int i = 0; i < 8; i++) {
        long r = rowBase + wm * 32 + mi * 16 + (lh ? (i + 8) : i);
        if (r < Nrows) {
          int c = colBase + wn * 64 + ni * 16 + lm;
          long idx = r * DDIM + c;
          Out[idx] = theta * acc[mi][ni][i] + omt * Sup[idx];
        }
      }
}

// ---------------------------------------------------------------------------
extern "C" void kernel_launch(void* const* d_in, const int* in_sizes, int n_in,
                              void* d_out, int out_size, void* d_ws, size_t ws_size,
                              hipStream_t stream) {
  const float* input  = (const float*)d_in[0];
  const int*   arow   = (const int*)d_in[1];
  const int*   acol   = (const int*)d_in[2];
  const float* aval   = (const float*)d_in[3];
  const float* h0     = (const float*)d_in[4];
  const float* weight = (const float*)d_in[5];
  const int*   lptr   = (const int*)d_in[6];
  int Nn = in_sizes[0] / DDIM;
  int E  = in_sizes[1];

  // workspace carve-up (floats)
  float* hi = (float*)d_ws;                       // N*D   (hi -> support)
  float* Zc = hi + (long)Nn * DDIM;               // 512*512
  float* S  = Zc + DDIM * DDIM;                   // 2*256*256
  float* B1 = S  + NGROUP * GD * GD;
  float* P  = B1 + NGROUP * GD * GD;
  float* Q  = P  + NGROUP * GD * GD;
  float* B2 = Q  + NGROUP * GD * GD;
  float* Tm = B2 + NGROUP * GD * GD;              // 512*512 (t)
  unsigned* Tp = (unsigned*)(Tm + DDIM * DDIM);   // packed f16 B frags
  float* gnorm = (float*)(Tp + DDIM * DDIM / 2);  // 4 floats (+pad)
  _Float16* Sup16 = (_Float16*)(gnorm + 8);       // N*D f16 copy of support

  long n4 = (long)Nn * DDIM / 4;
  const long sg = (long)GD * GD;                  // 65536
  const long sz = (long)GD * DDIM;                // 131072

  zero_k<<<2048, 256, 0, stream>>>((float4*)hi, n4);
  spmm_k<<<E, 256, 0, stream>>>(input, arow, acol, aval, hi);
  support_k<<<2048, 256, 0, stream>>>((float4*)hi, (const float4*)h0,
                                      (uint2*)Sup16, n4);
  zc_k<<<DDIM, 256, 0, stream>>>(weight, Zc);

  // S = Zc @ Zc^T + eps*I                 (NT, f32 WMMA)
  gemm32_k<1><<<dim3(8, 1, NGROUP), 256, 0, stream>>>(
      Zc, DDIM, sz, Zc, DDIM, sz, S, GD, sg,
      nullptr, 0, 0, nullptr, 0, GD, GD, DDIM, 1.0f, 0.0f, 1);
  norm_k<<<NGROUP, 256, 0, stream>>>(S, gnorm);
  snb1_k<<<(NGROUP * GD * GD + 255) / 256, 256, 0, stream>>>(S, B1, gnorm);

  // Newton-Schulz iteration 2 (B1, P, Q, Sn all symmetric -> NT loads)
  gemm32_k<1><<<dim3(8, 1, NGROUP), 256, 0, stream>>>(     // P = B1 @ B1
      B1, GD, sg, B1, GD, sg, P, GD, sg,
      nullptr, 0, 0, nullptr, 0, GD, GD, GD, 1.0f, 0.0f, 0);
  gemm32_k<1><<<dim3(8, 1, NGROUP), 256, 0, stream>>>(     // Q = P @ B1
      P, GD, sg, B1, GD, sg, Q, GD, sg,
      nullptr, 0, 0, nullptr, 0, GD, GD, GD, 1.0f, 0.0f, 0);
  gemm32_k<1><<<dim3(8, 1, NGROUP), 256, 0, stream>>>(     // B2 = 1.5*B1 - 0.5*Q@Sn
      Q, GD, sg, S, GD, sg, B2, GD, sg,
      B1, GD, sg, nullptr, 0, GD, GD, GD, -0.5f, 1.5f, 0);
  // t = (B2 @ Zc) * rsqrt(|S|)            (NN)
  gemm32_k<0><<<dim3(16, 1, NGROUP), 256, 0, stream>>>(
      B2, GD, sg, Zc, DDIM, sz, Tm, DDIM, sz,
      nullptr, 0, 0, gnorm + 1, 2, GD, DDIM, GD, 1.0f, 0.0f, 0);

  pack_k<<<(DDIM * DDIM / 2 + 255) / 256, 256, 0, stream>>>(Tm, Tp);

  dim3 grid((Nn + 63) / 64, DDIM / 256);
  biggemm_k<<<grid, 256, 0, stream>>>(hi, Sup16, Tp, lptr, (float*)d_out, Nn);
}
